// NonStationaryAdapter_60318520705655
// MI455X (gfx1250) — compile-verified, matrix operand order checked by reading, not run on hardware
//
#include <hip/hip_runtime.h>
#include <hip/hip_bf16.h>
#include <math.h>

typedef __attribute__((ext_vector_type(16))) _Float16 v16h;
typedef __attribute__((ext_vector_type(8)))  _Float16 v8h;
typedef __attribute__((ext_vector_type(8)))  float    v8f;
typedef __attribute__((ext_vector_type(4)))  unsigned int u32x4;
typedef __attribute__((ext_vector_type(8)))  int      i32x8;
typedef __attribute__((ext_vector_type(4)))  int      i32x4;

#define BATCH 8
#define SEQ   4096
#define DM    512
#define HD    256
#define ROWS  (BATCH * SEQ)          // 32768 (b,s) rows
#define NCHUNK 32                    // 4096 / 128
#define CHUNK  128
#define LDSTRIDE 520                 // 512 + 8 halves: +16B/row -> bank spread

// ---------------------------------------------------------------------------
// Fragment helpers for V_WMMA_F32_16X16X32_F16 (wave32).
// A (16x32 MxK): lane m (m=lane&15) holds row m; lanes 0-15 hold
// K=[0..7]+[16..23], lanes 16-31 hold K=[8..15]+[24..31].
// B (32x16 KxN) is symmetric with lane = column n; we compute X*W^T with W
// row-major (out,in), so lane n reads W row n contiguously in K.
// Works on generic pointers: global (global_load_b128) or LDS (ds_load_b128).
// ---------------------------------------------------------------------------
__device__ __forceinline__ v16h frag_ld(const _Float16* p) {
  v8h lo = *(const v8h*)p;
  v8h hi = *(const v8h*)(p + 16);
  v16h f;
#pragma unroll
  for (int i = 0; i < 8; ++i) { f[i] = lo[i]; f[i + 8] = hi[i]; }
  return f;
}

#define WMMA(A, B, C) __builtin_amdgcn_wmma_f32_16x16x32_f16( \
    false, A, false, B, (short)0, C, false, false)

#define LOADSET(S, KB)                                                        \
  a0##S = frag_ld(pa0 + (KB)); a1##S = frag_ld(pa1 + (KB));                   \
  b0##S = frag_ld(pb0 + (KB)); b1##S = frag_ld(pb1 + (KB));                   \
  b2##S = frag_ld(pb2 + (KB)); b3##S = frag_ld(pb3 + (KB));

#define MMASET(S)                                                             \
  acc00 = WMMA(a0##S, b0##S, acc00); acc10 = WMMA(a1##S, b0##S, acc10);       \
  acc01 = WMMA(a0##S, b1##S, acc01); acc11 = WMMA(a1##S, b1##S, acc11);       \
  acc02 = WMMA(a0##S, b2##S, acc02); acc12 = WMMA(a1##S, b2##S, acc12);       \
  acc03 = WMMA(a0##S, b3##S, acc03); acc13 = WMMA(a1##S, b3##S, acc13);

#define DECL_FRAGS                                                            \
  v16h a0A, a1A, b0A, b1A, b2A, b3A, a0B, a1B, b0B, b1B, b2B, b3B;            \
  v8f acc00, acc01, acc02, acc03, acc10, acc11, acc12, acc13;                 \
  {                                                                           \
    v8f z;                                                                    \
    _Pragma("unroll") for (int v = 0; v < 8; ++v) z[v] = 0.f;                 \
    acc00 = z; acc01 = z; acc02 = z; acc03 = z;                               \
    acc10 = z; acc11 = z; acc12 = z; acc13 = z;                               \
  }

// K=512 pipelined chain (16 K-steps), one stage of lookahead.
#define K512_PIPELINE                                                         \
  LOADSET(A, 0)                                                               \
  _Pragma("unroll") for (int kb = 0; kb < DM - 64; kb += 64) {                \
    LOADSET(B, kb + 32)                                                       \
    MMASET(A)                                                                 \
    LOADSET(A, kb + 64)                                                       \
    MMASET(B)                                                                 \
  }                                                                           \
  LOADSET(B, DM - 32)                                                         \
  MMASET(A)                                                                   \
  MMASET(B)

// ---------------------------------------------------------------------------
// Weight-slab staging: 64 rows x 512 halves (row stride in LDS = 520 halves).
// Preferred path: Tensor Data Mover (TENSORcnt). D# per ISA 08_async_tensor:
//   group0: count=1 | lds_addr | global_addr | type=2
//   group1: data_size=2B, pad_enable, pad_interval=256dw, pad_amount=4dw,
//           tensor_dim0=512, tensor_dim1=rows, tile_dim0=512, tile_dim1=rows,
//           tensor_dim0_stride=512
// This toolchain's builtin is the 6-arg form:
//   (u32x4 g0, i32x8 g1, i32x4 g2, i32x4 g3, i32x8 extra, i32 cpol)
// Fallback: cooperative global_load_b128 + ds_store_b128.
// ---------------------------------------------------------------------------
#if __has_builtin(__builtin_amdgcn_tensor_load_to_lds) && \
    __has_builtin(__builtin_amdgcn_s_wait_tensorcnt)
#define USE_TDM 1
#else
#define USE_TDM 0
#endif

__device__ __forceinline__ unsigned lds_offset_of(const void* p) {
  return (unsigned)(size_t)(__attribute__((address_space(3))) const void*)p;
}

__device__ __forceinline__ void stage_slab(const _Float16* gsrc, _Float16* lds) {
  __syncthreads();                       // prior consumers of this buffer done
#if USE_TDM
  if ((threadIdx.x >> 5) == 0) {
    const unsigned long long ga = (unsigned long long)(size_t)gsrc;
    u32x4 g0;
    g0[0] = 1u;                                          // count=1 (valid D#)
    g0[1] = lds_offset_of(lds);                          // lds_addr (bytes)
    g0[2] = (unsigned)(ga & 0xFFFFFFFFu);                // global_addr[31:0]
    g0[3] = (unsigned)((ga >> 32) & 0x01FFFFFFu) | 0x80000000u;  // [56:32]|type=2
    i32x8 g1;
    g1[0] = (1 << 16) | (1 << 20) | (7 << 22) | (3 << 25); // 2B|pad_en|256dw|4dw
    g1[1] = (int)(512u << 16);              // tensor_dim0[15:0]=512
    g1[2] = (int)(64u << 16);               // tensor_dim0 hi=0 | tensor_dim1 lo=64
    g1[3] = (int)(512u << 16);              // tensor_dim1 hi=0 | tile_dim0=512
    g1[4] = 64;                             // tile_dim1=64, tile_dim2=0
    g1[5] = 512;                            // tensor_dim0_stride lo32
    g1[6] = 0;
    g1[7] = 0;
    i32x4 z4;
    z4[0] = 0; z4[1] = 0; z4[2] = 0; z4[3] = 0;
    i32x8 z8;
#pragma unroll
    for (int i = 0; i < 8; ++i) z8[i] = 0;
    __builtin_amdgcn_tensor_load_to_lds(g0, g1, z4, z4, z8, 0);
    __builtin_amdgcn_s_wait_tensorcnt(0);
  }
#else
  for (unsigned c = threadIdx.x; c < 64u * 64u; c += 256u) {  // 16B chunks
    const unsigned row = c >> 6, col = c & 63u;
    *(v8h*)(lds + row * LDSTRIDE + col * 8) =
        *(const v8h*)(gsrc + (size_t)row * DM + col * 8);
  }
#endif
  __syncthreads();                       // slab visible to all waves
}

// ---------------------------------------------------------------------------
// fp32 -> fp16 conversions
// ---------------------------------------------------------------------------
__global__ __launch_bounds__(256) void k_convert_x(const float* __restrict__ x,
                                                   _Float16* __restrict__ xh, int n) {
  int i = blockIdx.x * 256 + threadIdx.x;
  if (i < n) xh[i] = (_Float16)x[i];
}

__global__ __launch_bounds__(256) void k_convert_w(const float* __restrict__ w1,
                                                   const float* __restrict__ l1,
                                                   const float* __restrict__ l2,
                                                   _Float16* __restrict__ w1h,
                                                   _Float16* __restrict__ l1h,
                                                   _Float16* __restrict__ l2h,
                                                   _Float16* __restrict__ zbuf) {
  int i = blockIdx.x * 256 + threadIdx.x;
  if (i < 3 * HD * DM) {
    int kk  = i / (HD * DM);
    int rem = i - kk * (HD * DM);
    w1h[i] = (_Float16)w1[rem * 3 + kk];
  }
  if (i < DM * DM) {
    l1h[i] = (_Float16)l1[i];
    l2h[i] = (_Float16)l2[i];
  }
  if (i < DM) zbuf[i] = (_Float16)0.f;
}

// ---------------------------------------------------------------------------
// conv1 as 3 shifted GEMMs (K=512 each): hcv = relu(X_shift*Wk^T + b)
// Block = 8 waves sharing one 64-col weight slab (TDM-staged per tap);
// each wave owns a 32x64 tile -> 384 WMMA/wave, B from LDS, A from global.
// ---------------------------------------------------------------------------
__global__ __launch_bounds__(256) void k_conv1(const _Float16* __restrict__ xh,
                                               const _Float16* __restrict__ w1h,
                                               const float* __restrict__ b1,
                                               const _Float16* __restrict__ zbuf,
                                               _Float16* __restrict__ hcvh) {
  __shared__ __align__(16) _Float16 ldsW[64 * LDSTRIDE];   // 66,560 B
  const int lane  = threadIdx.x & 31;
  const int w     = threadIdx.x >> 5;
  const int ng    = blockIdx.x & 3;
  const int mg    = blockIdx.x >> 2;        // 0..127
  const int mtile = mg * 8 + w;             // 0..1023 (32 rows each)
  const int nb0   = ng * 64;
  const int koff  = (lane & 16) ? 8 : 0;
  const int r0    = mtile * 32 + (lane & 15);
  const int r1    = r0 + 16;
  const int s0    = r0 & (SEQ - 1);
  const int s1    = s0 + 16;

  const _Float16* pzb = zbuf + koff;
  const _Float16* pb0 = ldsW + (size_t)(lane & 15) * LDSTRIDE + koff;
  const _Float16* pb1 = pb0 + 16 * LDSTRIDE;
  const _Float16* pb2 = pb0 + 32 * LDSTRIDE;
  const _Float16* pb3 = pb0 + 48 * LDSTRIDE;

  DECL_FRAGS

  for (int kk = 0; kk < 3; ++kk) {
    stage_slab(w1h + (size_t)kk * HD * DM + (size_t)nb0 * DM, ldsW);
    const int  sh  = kk - 1;
    const bool ok0 = (s0 + sh >= 0) && (s0 + sh < SEQ);
    const bool ok1 = (s1 + sh >= 0) && (s1 + sh < SEQ);
    const _Float16* pa0 = ok0 ? xh + (size_t)(r0 + sh) * DM + koff : pzb;
    const _Float16* pa1 = ok1 ? xh + (size_t)(r1 + sh) * DM + koff : pzb;
    { K512_PIPELINE }
  }

  const int colL = lane & 15;
  const int rb0  = mtile * 32 + ((lane >> 4) << 3);
  const int rb1  = rb0 + 16;
  const v8f accs0[4] = {acc00, acc01, acc02, acc03};
  const v8f accs1[4] = {acc10, acc11, acc12, acc13};
#pragma unroll
  for (int j = 0; j < 4; ++j) {
    const int   col  = nb0 + j * 16 + colL;
    const float bias = b1[col];
#pragma unroll
    for (int v = 0; v < 8; ++v) {
      float z0 = accs0[j][v] + bias;
      float z1 = accs1[j][v] + bias;
      hcvh[(size_t)(rb0 + v) * HD + col] = (_Float16)(z0 > 0.f ? z0 : 0.f);
      hcvh[(size_t)(rb1 + v) * HD + col] = (_Float16)(z1 > 0.f ? z1 : 0.f);
    }
  }
}

// ---------------------------------------------------------------------------
// conv2 (256->1, k=3) + sigmoid: one wave per (b,s), wave reduction.
// ---------------------------------------------------------------------------
__global__ __launch_bounds__(256) void k_conv2(const _Float16* __restrict__ hcvh,
                                               const float* __restrict__ w2,
                                               const float* __restrict__ b2c,
                                               float* __restrict__ change) {
  const int lane = threadIdx.x & 31;
  const int wave = blockIdx.x * 8 + (threadIdx.x >> 5);   // 0..32767
  const int b = wave >> 12, s = wave & (SEQ - 1);
  float sum = 0.f;
#pragma unroll
  for (int kk = 0; kk < 3; ++kk) {
    const int ss = s + kk - 1;
    if (ss < 0 || ss >= SEQ) continue;
    const _Float16* hp = hcvh + ((size_t)(b * SEQ + ss)) * HD;
#pragma unroll
    for (int h = lane; h < HD; h += 32)
      sum += (float)hp[h] * w2[h * 3 + kk];
  }
#pragma unroll
  for (int off = 16; off; off >>= 1) sum += __shfl_xor(sum, off, 32);
  if (lane == 0) change[wave] = 1.f / (1.f + __expf(-(sum + b2c[0])));
}

// ---------------------------------------------------------------------------
// MLP GEMM 1: m1 = tanh(X * W1^T + b1).  Block shares one TDM-staged slab.
// ---------------------------------------------------------------------------
__global__ __launch_bounds__(256) void k_mlp1(const _Float16* __restrict__ xh,
                                              const _Float16* __restrict__ l1h,
                                              const float* __restrict__ b1,
                                              _Float16* __restrict__ m1h) {
  __shared__ __align__(16) _Float16 ldsW[64 * LDSTRIDE];
  const int lane  = threadIdx.x & 31;
  const int w     = threadIdx.x >> 5;
  const int ng    = blockIdx.x & 7;
  const int mg    = blockIdx.x >> 3;        // 0..127
  const int mtile = mg * 8 + w;             // 0..1023
  const int nb0   = ng * 64;
  const int koff  = (lane & 16) ? 8 : 0;

  stage_slab(l1h + (size_t)nb0 * DM, ldsW);

  const _Float16* pa0 = xh + (size_t)(mtile * 32 + (lane & 15)) * DM + koff;
  const _Float16* pa1 = pa0 + (size_t)16 * DM;
  const _Float16* pb0 = ldsW + (size_t)(lane & 15) * LDSTRIDE + koff;
  const _Float16* pb1 = pb0 + 16 * LDSTRIDE;
  const _Float16* pb2 = pb0 + 32 * LDSTRIDE;
  const _Float16* pb3 = pb0 + 48 * LDSTRIDE;

  DECL_FRAGS
  K512_PIPELINE

  const int colL = lane & 15;
  const int rb0  = mtile * 32 + ((lane >> 4) << 3);
  const v8f accs0[4] = {acc00, acc01, acc02, acc03};
  const v8f accs1[4] = {acc10, acc11, acc12, acc13};
#pragma unroll
  for (int j = 0; j < 4; ++j) {
    const int   col  = nb0 + j * 16 + colL;
    const float bias = b1[col];
#pragma unroll
    for (int v = 0; v < 8; ++v) {
      m1h[(size_t)(rb0 + v) * DM + col]      = (_Float16)tanhf(accs0[j][v] + bias);
      m1h[(size_t)(rb0 + 16 + v) * DM + col] = (_Float16)tanhf(accs1[j][v] + bias);
    }
  }
}

// ---------------------------------------------------------------------------
// MLP GEMM 2 + blend: m = m1*W2^T + b2 ; bvec = c*m + (1-c)(1-sw)*x
// s==0 rows carry y_0 = x_0 (a_0 forced to 0 keeps the scan uniform).
// ---------------------------------------------------------------------------
__global__ __launch_bounds__(256) void k_mlp2(const _Float16* __restrict__ m1h,
                                              const _Float16* __restrict__ l2h,
                                              const float* __restrict__ b2,
                                              const float* __restrict__ x32,
                                              const float* __restrict__ change,
                                              const float* __restrict__ swp,
                                              float* __restrict__ adapted) {
  __shared__ __align__(16) _Float16 ldsW[64 * LDSTRIDE];
  const int lane  = threadIdx.x & 31;
  const int w     = threadIdx.x >> 5;
  const int ng    = blockIdx.x & 7;
  const int mg    = blockIdx.x >> 3;
  const int mtile = mg * 8 + w;
  const int nb0   = ng * 64;
  const int koff  = (lane & 16) ? 8 : 0;

  stage_slab(l2h + (size_t)nb0 * DM, ldsW);

  const _Float16* pa0 = m1h + (size_t)(mtile * 32 + (lane & 15)) * DM + koff;
  const _Float16* pa1 = pa0 + (size_t)16 * DM;
  const _Float16* pb0 = ldsW + (size_t)(lane & 15) * LDSTRIDE + koff;
  const _Float16* pb1 = pb0 + 16 * LDSTRIDE;
  const _Float16* pb2 = pb0 + 32 * LDSTRIDE;
  const _Float16* pb3 = pb0 + 48 * LDSTRIDE;

  DECL_FRAGS
  K512_PIPELINE

  const float sw   = 1.f / (1.f + __expf(-swp[0]));
  const int   colL = lane & 15;
  const int   rb0  = mtile * 32 + ((lane >> 4) << 3);
  const v8f accs0[4] = {acc00, acc01, acc02, acc03};
  const v8f accs1[4] = {acc10, acc11, acc12, acc13};
#pragma unroll
  for (int j = 0; j < 4; ++j) {
    const int   col  = nb0 + j * 16 + colL;
    const float bias = b2[col];
#pragma unroll
    for (int v = 0; v < 8; ++v) {
#pragma unroll
      for (int half = 0; half < 2; ++half) {
        const int   row = rb0 + half * 16 + v;
        const float m   = (half ? accs1[j][v] : accs0[j][v]) + bias;
        const float c   = change[row];
        const float xv  = x32[(size_t)row * DM + col];
        const float bv  = ((row & (SEQ - 1)) == 0)
                              ? xv
                              : (c * m + (1.f - c) * (1.f - sw) * xv);
        adapted[(size_t)row * DM + col] = bv;
      }
    }
  }
}

// a_t = (1-c_t)*sw for t>=1 ; a_0 = 0 (folds y_0 = x_0 into the scan)
__global__ __launch_bounds__(256) void k_avec(const float* __restrict__ change,
                                              const float* __restrict__ swp,
                                              float* __restrict__ a, int n) {
  int i = blockIdx.x * 256 + threadIdx.x;
  if (i < n) {
    const float sw = 1.f / (1.f + __expf(-swp[0]));
    a[i] = ((i & (SEQ - 1)) == 0) ? 0.f : (1.f - change[i]) * sw;
  }
}

// ---------------------------------------------------------------------------
// Chunked scan, phase 1: local scan of 128 steps (carry-in 0), in place over
// adapted (holds bvec), plus per-t prefix products p and chunk-end vectors.
// ---------------------------------------------------------------------------
__global__ __launch_bounds__(512) void k_scan_local(const float* __restrict__ a,
                                                    float* __restrict__ adapted,
                                                    float* __restrict__ p,
                                                    float* __restrict__ yend) {
  const int chunk = blockIdx.x;           // 0..255
  const int b = chunk >> 5, k = chunk & (NCHUNK - 1);
  const int t0 = k * CHUNK;
  __shared__ float as[CHUNK];
  if (threadIdx.x < CHUNK) as[threadIdx.x] = a[b * SEQ + t0 + threadIdx.x];
  __syncthreads();
  if (threadIdx.x == 0) {                 // per-t prefix products (scalar chain)
    float pp = 1.f;
#pragma unroll 8
    for (int i = 0; i < CHUNK; ++i) { pp *= as[i]; p[b * SEQ + t0 + i] = pp; }
  }
  const int d = threadIdx.x;
  float y = 0.f;
  size_t base = ((size_t)b * SEQ + t0) * DM + d;
#pragma unroll 4
  for (int i = 0; i < CHUNK; ++i) {
    y = as[i] * y + adapted[base + (size_t)i * DM];
    adapted[base + (size_t)i * DM] = y;
  }
  yend[(size_t)chunk * DM + d] = y;
}

// phase 2: sequential combine across 32 chunks per batch (vector carries)
__global__ __launch_bounds__(512) void k_scan_combine(const float* __restrict__ p,
                                                      const float* __restrict__ yend,
                                                      float* __restrict__ cin) {
  const int b = blockIdx.x, d = threadIdx.x;
  float y = 0.f;
#pragma unroll
  for (int k = 0; k < NCHUNK; ++k) {
    const size_t idx = ((size_t)(b * NCHUNK + k)) * DM + d;
    cin[idx] = y;
    const float P = p[b * SEQ + k * CHUNK + (CHUNK - 1)];
    y = yend[idx] + P * y;
  }
}

// phase 3: y_t += p_t * carry_in(chunk)
__global__ __launch_bounds__(512) void k_scan_fixup(float* __restrict__ adapted,
                                                    const float* __restrict__ p,
                                                    const float* __restrict__ cin) {
  const size_t idx = (size_t)blockIdx.x * 512 + threadIdx.x;   // < ROWS*DM
  const size_t r = idx >> 9;                                   // row (b,t)
  adapted[idx] += p[r] * cin[(idx >> 16) * DM + (idx & (DM - 1))];
}

// ---------------------------------------------------------------------------
extern "C" void kernel_launch(void* const* d_in, const int* in_sizes, int n_in,
                              void* d_out, int out_size, void* d_ws, size_t ws_size,
                              hipStream_t stream) {
  const float* x   = (const float*)d_in[0];
  const float* c1w = (const float*)d_in[1];
  const float* c1b = (const float*)d_in[2];
  const float* c2w = (const float*)d_in[3];
  const float* c2b = (const float*)d_in[4];
  const float* l1w = (const float*)d_in[5];
  const float* l1b = (const float*)d_in[6];
  const float* l2w = (const float*)d_in[7];
  const float* l2b = (const float*)d_in[8];
  const float* swp = (const float*)d_in[9];

  float* adapted = (float*)d_out;                     // (B,S,D)
  float* change  = adapted + (size_t)ROWS * DM;       // (B,S)

  char* ws = (char*)d_ws;
  size_t o = 0;
  auto carve = [&](size_t bytes) { size_t r = o; o += (bytes + 255) & ~(size_t)255; return r; };
  _Float16* xh   = (_Float16*)(ws + carve((size_t)ROWS * DM * 2));   // 32 MB
  _Float16* m1h  = (_Float16*)(ws + carve((size_t)ROWS * DM * 2));   // 32 MB
  _Float16* hcvh = (_Float16*)(ws + carve((size_t)ROWS * HD * 2));   // 16 MB
  _Float16* w1h  = (_Float16*)(ws + carve((size_t)3 * HD * DM * 2));
  _Float16* l1h  = (_Float16*)(ws + carve((size_t)DM * DM * 2));
  _Float16* l2h  = (_Float16*)(ws + carve((size_t)DM * DM * 2));
  _Float16* zbuf = (_Float16*)(ws + carve((size_t)DM * 2));
  float*    avec = (float*)(ws + carve((size_t)ROWS * 4));
  float*    pbuf = (float*)(ws + carve((size_t)ROWS * 4));
  float*    yend = (float*)(ws + carve((size_t)BATCH * NCHUNK * DM * 4));
  float*    cin  = (float*)(ws + carve((size_t)BATCH * NCHUNK * DM * 4));

  const int nX = ROWS * DM;
  k_convert_x<<<nX / 256, 256, 0, stream>>>(x, xh, nX);
  k_convert_w<<<(3 * HD * DM) / 256, 256, 0, stream>>>(c1w, l1w, l2w, w1h, l1h, l2h, zbuf);

  // conv1: 4 ngroups x 128 mgroups = 512 blocks, 8 waves each
  k_conv1<<<512, 256, 0, stream>>>(xh, w1h, c1b, zbuf, hcvh);
  k_conv2<<<ROWS / 8, 256, 0, stream>>>(hcvh, c2w, c2b, change);
  // mlp: 8 ngroups x 128 mgroups = 1024 blocks
  k_mlp1<<<1024, 256, 0, stream>>>(xh, l1h, l1b, m1h);
  k_avec<<<ROWS / 256, 256, 0, stream>>>(change, swp, avec, ROWS);
  k_mlp2<<<1024, 256, 0, stream>>>(m1h, l2h, l2b, x, change, swp, adapted);

  k_scan_local<<<BATCH * NCHUNK, 512, 0, stream>>>(avec, adapted, pbuf, yend);
  k_scan_combine<<<BATCH, 512, 0, stream>>>(pbuf, yend, cin);
  k_scan_fixup<<<nX / 512, 512, 0, stream>>>(adapted, pbuf, cin);
}
